// TwistorLNN_5660766896353
// MI455X (gfx1250) — matrix operand order repaired and checked by reading, not compiled, not run on hardware
//
#include <hip/hip_runtime.h>

typedef __attribute__((ext_vector_type(16))) _Float16 v16h;
typedef __attribute__((ext_vector_type(8)))  float    v8f;
typedef __attribute__((ext_vector_type(4)))  float    v4f;

#define T_STEPS 2048
#define BATCH   512
#define DIN     64
#define HID     16
#define LOG2E   1.4426950408889634f
#define INV2PI  0.15915494309189535f

// Compiler-only ordering fence. Per CDNA5 ISA §7.1/7.3, DS instructions from a
// single wave are processed in order (store -> load visibility within a wave is
// guaranteed in hardware). With one wave per block, no s_barrier / s_wait is
// needed between the LDS transpose store and load — only the compiler must not
// reorder them. This avoids the s_wait_loadcnt_dscnt 0x0 that __syncthreads()
// emitted, which was draining the independent global x-loads into the chain.
__device__ __forceinline__ void lds_order_fence() {
  asm volatile("" ::: "memory");
}

__device__ __forceinline__ v16h pack_row16(const float* __restrict__ p, float s) {
  v16h r;
#pragma unroll
  for (int i = 0; i < 16; ++i) r[i] = (_Float16)(p[i] * s);
  return r;
}

__device__ __forceinline__ v16h zero16h() {
  v16h r;
#pragma unroll
  for (int i = 0; i < 16; ++i) r[i] = (_Float16)0.0f;
  return r;
}

__global__ __launch_bounds__(32) void twistor_scan_kernel(
    const float* __restrict__ x,
    const float* __restrict__ W_re, const float* __restrict__ W_im,
    const float* __restrict__ U_re, const float* __restrict__ U_im,
    const float* __restrict__ b_re, const float* __restrict__ b_im,
    const float* __restrict__ W_tau, const float* __restrict__ b_tau,
    const float* __restrict__ W_out, const float* __restrict__ b_out,
    float* __restrict__ y)
{
  const int lane = threadIdx.x;     // wave32: one wave per block
  const int n    = lane & 15;       // column (N / H index) owned in C/D layout
  const int hi   = lane >> 4;       // half-wave select
  const int b0   = blockIdx.x * 16; // batch tile base

  // ---------------- B-operand fragments (constant over T) ----------------
  // B packing used here (K order matches A): lanes 0-15 hold column N=lane, K=0..15; lanes 16-31
  // hold column N=lane-16, K=16..31. Complex pack along K:
  //   C_re uses B=[W_re ; -W_im],  C_im uses B=[W_im ; W_re].
  v16h bWcr = (hi == 0) ? pack_row16(W_re  + n * HID,  1.f)
                        : pack_row16(W_im  + n * HID, -1.f);
  v16h bWci = (hi == 0) ? pack_row16(W_im  + n * HID,  1.f)
                        : pack_row16(W_re  + n * HID,  1.f);
  v16h bTau = (hi == 0) ? pack_row16(W_tau + n * HID,  1.f) : zero16h();
  v16h bUr0 = pack_row16(U_re + n * DIN +      hi * 16, 1.f);
  v16h bUr1 = pack_row16(U_re + n * DIN + 32 + hi * 16, 1.f);
  v16h bUi0 = pack_row16(U_im + n * DIN +      hi * 16, 1.f);
  v16h bUi1 = pack_row16(U_im + n * DIN + 32 + hi * 16, 1.f);

  const float brn    = b_re[n];
  const float bin    = b_im[n];
  const float btn    = b_tau[n];
  const float wout_n = W_out[n];
  const float bout   = b_out[0];

  // ---------------- persistent complex state in accumulator layout --------
  v8f zr, zi;
#pragma unroll
  for (int j = 0; j < 8; ++j) { zr[j] = 0.f; zi[j] = 0.f; }

  // LDS staging tiles (accumulator layout -> A-operand layout transpose).
  // Row stride 40 halves (80B): the 16 row-reads land on distinct banks.
  __shared__ _Float16 ldsC[16][40];  // [tanh_re | tanh_im] along K (0..31)
  __shared__ _Float16 ldsA[16][40];  // [|z| | zeros]       along K (0..31)
  for (int i = lane; i < 16 * 40; i += 32) {
    (&ldsC[0][0])[i] = (_Float16)0.f;
    (&ldsA[0][0])[i] = (_Float16)0.f;
  }
  __syncthreads();  // one-time init barrier only

  const int M  = n;            // A-fragment row owned by this lane
  const int Kb = hi ? 16 : 0;  // A-fragment K base for this half-wave

  // ---------------- software-pipelined x loads ----------------------------
  // xq holds raw f32 of row M for step t+1; axc0/axc1 are the converted f16
  // A-fragments for the *current* step.
  v4f xq[8];
  {
    const float* __restrict__ xr0 = x + ((size_t)0 * BATCH + b0 + M) * DIN;
    xq[0] = *(const v4f*)(xr0 + Kb);       xq[1] = *(const v4f*)(xr0 + Kb + 4);
    xq[2] = *(const v4f*)(xr0 + Kb + 8);   xq[3] = *(const v4f*)(xr0 + Kb + 12);
    xq[4] = *(const v4f*)(xr0 + Kb + 32);  xq[5] = *(const v4f*)(xr0 + Kb + 36);
    xq[6] = *(const v4f*)(xr0 + Kb + 40);  xq[7] = *(const v4f*)(xr0 + Kb + 44);
  }
  v16h axc0, axc1;
#pragma unroll
  for (int i = 0; i < 4; ++i) {
    axc0[i]      = (_Float16)xq[0][i];  axc0[4 + i]  = (_Float16)xq[1][i];
    axc0[8 + i]  = (_Float16)xq[2][i];  axc0[12 + i] = (_Float16)xq[3][i];
    axc1[i]      = (_Float16)xq[4][i];  axc1[4 + i]  = (_Float16)xq[5][i];
    axc1[8 + i]  = (_Float16)xq[6][i];  axc1[12 + i] = (_Float16)xq[7][i];
  }

  for (int t = 0; t < T_STEPS; ++t) {
    // ---- issue global loads for step t+1 (latency hidden over whole step)
    const int tn = (t + 1 < T_STEPS) ? t + 1 : T_STEPS - 1;
    const float* __restrict__ xr = x + ((size_t)tn * BATCH + b0 + M) * DIN;
    xq[0] = *(const v4f*)(xr + Kb);       xq[1] = *(const v4f*)(xr + Kb + 4);
    xq[2] = *(const v4f*)(xr + Kb + 8);   xq[3] = *(const v4f*)(xr + Kb + 12);
    xq[4] = *(const v4f*)(xr + Kb + 32);  xq[5] = *(const v4f*)(xr + Kb + 36);
    xq[6] = *(const v4f*)(xr + Kb + 40);  xq[7] = *(const v4f*)(xr + Kb + 44);

    // ---- input drive Ux for current step: fragments ready, XDL co-executes
    // with the transcendental VALU chain below.
    v8f uxr, uxi;
#pragma unroll
    for (int j = 0; j < 8; ++j) { uxr[j] = 0.f; uxi[j] = 0.f; }
    uxr = __builtin_amdgcn_wmma_f32_16x16x32_f16(false, axc0, false, bUr0, (short)0, uxr, false, false);
    uxr = __builtin_amdgcn_wmma_f32_16x16x32_f16(false, axc1, false, bUr1, (short)0, uxr, false, false);
    uxi = __builtin_amdgcn_wmma_f32_16x16x32_f16(false, axc0, false, bUi0, (short)0, uxi, false, false);
    uxi = __builtin_amdgcn_wmma_f32_16x16x32_f16(false, axc1, false, bUi1, (short)0, uxi, false, false);

    // ---- complex tanh(z) and safe |z| -> LDS (division-free fast path) ----
#pragma unroll
    for (int j = 0; j < 8; ++j) {
      const int m = j + hi * 8;
      const float a = zr[j], b = zi[j];
      // tanh(a) = (e-1)/(e+1), e = 2^(2a*log2e); clamp avoids inf*0
      const float ac = fminf(fmaxf(a, -15.f), 15.f);
      const float ea = __builtin_amdgcn_exp2f(ac * (2.f * LOG2E));
      const float ta = (ea - 1.f) * __builtin_amdgcn_rcpf(ea + 1.f);
      // tan(b) via hardware sin/cos (args pre-scaled by 1/2pi)
      const float sb = __builtin_amdgcn_sinf(b * INV2PI);
      const float cb = __builtin_amdgcn_cosf(b * INV2PI);
      const float tb = sb * __builtin_amdgcn_rcpf(cb);
      const float a2 = ta * ta, t2 = tb * tb;
      const float rd = __builtin_amdgcn_rcpf(1.f + a2 * t2);
      ldsC[m][n]      = (_Float16)(ta * (1.f + t2) * rd);  // Re tanh(a+ib)
      ldsC[m][n + 16] = (_Float16)(tb * (1.f - a2) * rd);  // Im tanh(a+ib)
      ldsA[m][n]      = (_Float16)__builtin_amdgcn_sqrtf(a * a + b * b + 1e-12f);
    }
    lds_order_fence();   // in-order per-wave DS: compiler ordering is enough

    // ---- gather A fragments (elements 0..15 = K[Kb..Kb+15])
    v16h aC, aA;
#pragma unroll
    for (int j = 0; j < 8; ++j) {
      aC[j]     = ldsC[M][Kb + j];
      aC[8 + j] = ldsC[M][Kb + 8 + j];
      aA[j]     = ldsA[M][Kb + j];
      aA[8 + j] = ldsA[M][Kb + 8 + j];
    }
    lds_order_fence();   // keep next iteration's stores behind these loads

    v8f c0;
#pragma unroll
    for (int j = 0; j < 8; ++j) c0[j] = 0.f;
    // complex GEMM in two WMMAs via K-packing + tau GEMM
    v8f wr = __builtin_amdgcn_wmma_f32_16x16x32_f16(false, aC, false, bWcr, (short)0, c0, false, false);
    v8f wi = __builtin_amdgcn_wmma_f32_16x16x32_f16(false, aC, false, bWci, (short)0, c0, false, false);
    v8f ts = __builtin_amdgcn_wmma_f32_16x16x32_f16(false, aA, false, bTau, (short)0, c0, false, false);

    // ---- state update (division-free) + readout partials ----
    float part[8];
#pragma unroll
    for (int j = 0; j < 8; ++j) {
      const float s = ts[j] + btn;
      // 1/clamp(sigmoid(s),0.01,1) == clamp(1+e^{-s},1,100): no division
      const float e2   = __builtin_amdgcn_exp2f(-s * LOG2E);
      const float itau = fminf(fmaxf(1.f + e2, 1.f), 100.f);
      const float dr = (wr[j] + uxr[j] + (brn - zr[j])) * itau;
      const float di = (wi[j] + uxi[j] + (bin - zi[j])) * itau;
      const float m2 = dr * dr + di * di + 1e-12f;
      float sc = fminf(1.f, 10.f * __builtin_amdgcn_rsqf(m2)); // |dz| <= 10
      sc *= 0.1f;                                              // fold DT
      const float a = zr[j] + dr * sc;
      const float b = zi[j] + di * sc;
      zr[j] = fminf(fmaxf(a, -100.f), 100.f);
      zi[j] = fminf(fmaxf(b, -100.f), 100.f);
      part[j] = zr[j] * wout_n;
    }
    // exact f32 butterfly reduction across the 16 lanes of each half-wave
#pragma unroll
    for (int j = 0; j < 8; ++j) {
#pragma unroll
      for (int mask = 1; mask < 16; mask <<= 1)
        part[j] += __shfl_xor(part[j], mask, 32);
    }
    if (n == 0) {  // lanes 0 (rows 0-7) and 16 (rows 8-15)
      float* __restrict__ yrow = y + (size_t)t * BATCH + b0 + hi * 8;
#pragma unroll
      for (int j = 0; j < 8; ++j) yrow[j] = part[j] + bout;
    }

    // ---- convert t+1 x row to f16 A-fragments (waits on loads issued above)
#pragma unroll
    for (int i = 0; i < 4; ++i) {
      axc0[i]      = (_Float16)xq[0][i];  axc0[4 + i]  = (_Float16)xq[1][i];
      axc0[8 + i]  = (_Float16)xq[2][i];  axc0[12 + i] = (_Float16)xq[3][i];
      axc1[i]      = (_Float16)xq[4][i];  axc1[4 + i]  = (_Float16)xq[5][i];
      axc1[8 + i]  = (_Float16)xq[6][i];  axc1[12 + i] = (_Float16)xq[7][i];
    }
  }
}

extern "C" void kernel_launch(void* const* d_in, const int* in_sizes, int n_in,
                              void* d_out, int out_size, void* d_ws, size_t ws_size,
                              hipStream_t stream) {
  (void)in_sizes; (void)n_in; (void)d_ws; (void)ws_size; (void)out_size;
  const float* x     = (const float*)d_in[0];
  const float* W_re  = (const float*)d_in[1];
  const float* W_im  = (const float*)d_in[2];
  const float* U_re  = (const float*)d_in[3];
  const float* U_im  = (const float*)d_in[4];
  const float* b_re  = (const float*)d_in[5];
  const float* b_im  = (const float*)d_in[6];
  const float* W_tau = (const float*)d_in[7];
  const float* b_tau = (const float*)d_in[8];
  const float* W_out = (const float*)d_in[9];
  const float* b_out = (const float*)d_in[10];
  float* y = (float*)d_out;

  // 32 persistent waves: one 16-row batch tile each, full T-scan per wave.
  twistor_scan_kernel<<<dim3(BATCH / 16), dim3(32), 0, stream>>>(
      x, W_re, W_im, U_re, U_im, b_re, b_im, W_tau, b_tau, W_out, b_out, y);
}
